// SampledSoftmaxLoss_32109175505609
// MI455X (gfx1250) — matrix operand order repaired and checked by reading, not compiled, not run on hardware
//
#include <hip/hip_runtime.h>
#include <hip/hip_bf16.h>

// ---------------- problem constants (match the reference) ----------------
#define B_DIM 8192
#define D_DIM 512
#define S_DIM 8192

#define NCHUNK 8                         // column chunks across S (grid.x)
#define COLS_PER_CHUNK (S_DIM / NCHUNK)  // 1024
#define BM 64
#define BN 128
#define BK 32
#define KSTEPS (D_DIM / BK)              // 16
#define COLSTEPS (COLS_PER_CHUNK / BN)   // 8
#define LDK 36                           // padded bf16 K-stride (dword-even, conflict-free)
#define LDL 68                           // padded logits row stride (floats)

typedef __attribute__((ext_vector_type(16))) __bf16        v16bf;
typedef __attribute__((ext_vector_type(8)))  float         v8f;
typedef __attribute__((ext_vector_type(8)))  unsigned int  v8u;

// ---------------- helpers ----------------
__device__ __forceinline__ unsigned short f32_to_bf16_rne(float f) {
    unsigned int u = __float_as_uint(f);
    u += 0x7FFFu + ((u >> 16) & 1u);     // round-to-nearest-even
    return (unsigned short)(u >> 16);
}
__device__ __forceinline__ float bf16_to_f32(unsigned short h) {
    return __uint_as_float(((unsigned int)h) << 16);
}
__device__ __forceinline__ unsigned int pack_bf16_hi(float x, float y) {
    return (unsigned int)f32_to_bf16_rne(x) | ((unsigned int)f32_to_bf16_rne(y) << 16);
}

// A fragment (16x32 bf16): lanes 0-15 -> M=lane, K 0..15 ; lanes 16-31 -> K 16..31
__device__ __forceinline__ v16bf load_frag_a(const unsigned short* base, int mBase, int lane) {
    const int m  = mBase + (lane & 15);
    const int kq = (lane >> 4) * 8;                  // dword offset within row
    const unsigned int* p = (const unsigned int*)(base + m * LDK);
    v8u r;
#pragma unroll
    for (int v = 0; v < 4; ++v) { r[v] = p[kq + v]; r[4 + v] = p[kq + 4 + v]; }
    return __builtin_bit_cast(v16bf, r);
}
// B fragment (32x16 bf16): lanes 0-15 -> N=lane, K 0..15 ; lanes 16-31 -> K 16..31
__device__ __forceinline__ v16bf load_frag_b(const unsigned short* base, int nBase, int lane) {
    const int n  = nBase + (lane & 15);
    const int kq = (lane >> 4) * 8;
    const unsigned int* p = (const unsigned int*)(base + n * LDK);
    v8u r;
#pragma unroll
    for (int v = 0; v < 8; ++v) r[v] = p[kq + v];
    return __builtin_bit_cast(v16bf, r);
}
__device__ __forceinline__ v8f wmma_bf16(v16bf a, v16bf b, v8f c) {
    return __builtin_amdgcn_wmma_f32_16x16x32_bf16(false, a, false, b, (short)0, c, false, false);
}
__device__ __forceinline__ float neg_log_expected_count(int id, float num_tries, float logVp1) {
    // p = (log(id+2)-log(id+1)) / log(V+1) ;  cnt = -expm1(num_tries*log1p(-p))
    float idf = (float)id;
    float p   = (logf(idf + 2.0f) - logf(idf + 1.0f)) / logVp1;
    float cnt = -expm1f(num_tries * log1pf(-p));
    return -logf(cnt + 1e-7f);
}

// ---------------- kernel 1: per-sampled-column constants ----------------
__global__ __launch_bounds__(256) void col_const_kernel(
    const int* __restrict__ sampled_ids, const float* __restrict__ softmax_b,
    const int* __restrict__ num_tries_p, int V, float* __restrict__ colconst) {
    int s = blockIdx.x * 256 + threadIdx.x;
    if (s >= S_DIM) return;
    int id = sampled_ids[s];
    float nt = (float)num_tries_p[0];
    float logVp1 = logf((float)V + 1.0f);
    colconst[s] = softmax_b[id] + neg_log_expected_count(id, nt, logVp1);
}

// ---------------- kernel 2: true logits (per-row D-dot + bias terms) ----------------
__global__ __launch_bounds__(256) void true_logit_kernel(
    const float* __restrict__ emb, const int* __restrict__ targets,
    const float* __restrict__ w, const float* __restrict__ softmax_b,
    const int* __restrict__ num_tries_p, int V, float* __restrict__ true_logits) {
    __shared__ float red[256];
    const int row = blockIdx.x;
    const int t   = targets[row];
    const float* e  = emb + (size_t)row * D_DIM;
    const float* wr = w   + (size_t)t   * D_DIM;
    float s = 0.0f;
    for (int i = threadIdx.x; i < D_DIM; i += 256) s += e[i] * wr[i];
    red[threadIdx.x] = s;
    __syncthreads();
    for (int off = 128; off > 0; off >>= 1) {
        if ((int)threadIdx.x < off) red[threadIdx.x] += red[threadIdx.x + off];
        __syncthreads();
    }
    if (threadIdx.x == 0) {
        float nt = (float)num_tries_p[0];
        float logVp1 = logf((float)V + 1.0f);
        true_logits[row] = red[0] + softmax_b[t] + neg_log_expected_count(t, nt, logVp1);
    }
}

// ---------------- kernel 3: fused bf16x3 WMMA GEMM + online softmax partials ----------------
__global__ __launch_bounds__(256) void gemm_lse_kernel(
    const float* __restrict__ emb, const int* __restrict__ targets,
    const int* __restrict__ sampled_ids, const float* __restrict__ w,
    const float* __restrict__ colconst, float* __restrict__ partials) {
    __shared__ unsigned short sAh[BM][LDK], sAl[BM][LDK];
    __shared__ unsigned short sBh[BN][LDK], sBl[BN][LDK];
    __shared__ float sLog[BM][LDL];          // 64 x 64 logits (one half of BN) per pass
    __shared__ int   sIds[BN];
    __shared__ float sCc[BN];
    __shared__ int   sTgt[BM];
    __shared__ float sRunM[BM], sRunL[BM];
    __shared__ float sPm[BM][4], sPl[BM][4];

    const int tid  = threadIdx.x;
    const int lane = tid & 31;
    const int wave = tid >> 5;        // 0..7
    const int wrow = wave >> 2;       // 0..1 : owns rows wrow*32 .. +32
    const int wcol = wave & 3;        // 0..3 : owns cols wcol*32 .. +32
    const int chunk   = blockIdx.x;   // 0..7
    const int rowBase = blockIdx.y * BM;
    const int chunkColBase = chunk * COLS_PER_CHUNK;

    if (tid < BM) {
        sTgt[tid]  = targets[rowBase + tid];
        sRunM[tid] = -3.0e38f;
        sRunL[tid] = 0.0f;
    }
    __syncthreads();

    for (int cs = 0; cs < COLSTEPS; ++cs) {
        const int colBase = chunkColBase + cs * BN;
        if (tid < BN) {
            sIds[tid] = sampled_ids[colBase + tid];
            sCc[tid]  = colconst[colBase + tid];
        }
        v8f acc[2][2];
#pragma unroll
        for (int mi = 0; mi < 2; ++mi)
#pragma unroll
            for (int ni = 0; ni < 2; ++ni)
                acc[mi][ni] = (v8f){0.f, 0.f, 0.f, 0.f, 0.f, 0.f, 0.f, 0.f};

        for (int kk = 0; kk < KSTEPS; ++kk) {
            __syncthreads();                       // also fences sIds on first iter
            const int k0 = kk * BK;
            // ---- stage A (64x32 fp32 -> packed bf16 hi/lo) : 4 pairs / thread ----
#pragma unroll
            for (int i = 0; i < 4; ++i) {
                int p  = tid + i * 256;            // pair index, 1024 total
                int r  = p >> 4;                   // 0..63
                int c2 = (p & 15) * 2;             // 0..30, even
                float2 f2 = *(const float2*)(emb + (size_t)(rowBase + r) * D_DIM + k0 + c2);
                float hx = bf16_to_f32(f32_to_bf16_rne(f2.x));
                float hy = bf16_to_f32(f32_to_bf16_rne(f2.y));
                *(unsigned int*)(&sAh[r][c2]) = pack_bf16_hi(f2.x, f2.y);
                *(unsigned int*)(&sAl[r][c2]) = pack_bf16_hi(f2.x - hx, f2.y - hy);
            }
            // ---- stage B gathered rows (128x32 fp32 -> packed bf16 hi/lo) : 8 pairs / thread ----
#pragma unroll
            for (int i = 0; i < 8; ++i) {
                int p  = tid + i * 256;            // pair index, 2048 total
                int r  = p >> 4;                   // 0..127
                int c2 = (p & 15) * 2;
                float2 f2 = *(const float2*)(w + (size_t)sIds[r] * D_DIM + k0 + c2);
                float hx = bf16_to_f32(f32_to_bf16_rne(f2.x));
                float hy = bf16_to_f32(f32_to_bf16_rne(f2.y));
                *(unsigned int*)(&sBh[r][c2]) = pack_bf16_hi(f2.x, f2.y);
                *(unsigned int*)(&sBl[r][c2]) = pack_bf16_hi(f2.x - hx, f2.y - hy);
            }
            __syncthreads();
            // ---- fragments + bf16x3 WMMA : 8 loads -> 12 WMMAs ----
            v16bf aH[2], aL[2], bH[2], bL[2];
#pragma unroll
            for (int mi = 0; mi < 2; ++mi) {
                aH[mi] = load_frag_a(&sAh[0][0], wrow * 32 + mi * 16, lane);
                aL[mi] = load_frag_a(&sAl[0][0], wrow * 32 + mi * 16, lane);
            }
#pragma unroll
            for (int ni = 0; ni < 2; ++ni) {
                bH[ni] = load_frag_b(&sBh[0][0], wcol * 32 + ni * 16, lane);
                bL[ni] = load_frag_b(&sBl[0][0], wcol * 32 + ni * 16, lane);
            }
#pragma unroll
            for (int mi = 0; mi < 2; ++mi)
#pragma unroll
                for (int ni = 0; ni < 2; ++ni) {
                    acc[mi][ni] = wmma_bf16(aH[mi], bH[ni], acc[mi][ni]);
                    acc[mi][ni] = wmma_bf16(aH[mi], bL[ni], acc[mi][ni]);
                    acc[mi][ni] = wmma_bf16(aL[mi], bH[ni], acc[mi][ni]);
                }
        }

        // ---- epilogue in two 64-column passes (keeps LDS logits tile small) ----
#pragma unroll
        for (int pass = 0; pass < 2; ++pass) {
            if ((wcol >> 1) == pass) {
                // C layout: lanes 0-15: N=lane, M=v ; lanes 16-31: N=lane-16, M=8+v
                const int n16     = lane & 15;
                const int moBase  = wrow * 32 + ((lane >> 4) << 3);
                const int nLocB   = (wcol & 1) * 32;
#pragma unroll
                for (int mi = 0; mi < 2; ++mi)
#pragma unroll
                    for (int ni = 0; ni < 2; ++ni) {
                        const int nLoc = nLocB + ni * 16 + n16;   // 0..63 within pass
                        const int nIdx = pass * 64 + nLoc;        // 0..127 within col step
                        const float cc = sCc[nIdx];
                        const int   ci = sIds[nIdx];
#pragma unroll
                        for (int v = 0; v < 8; ++v) {
                            const int m = moBase + mi * 16 + v;
                            float val = acc[mi][ni][v] + cc;
                            if (ci == sTgt[m]) val = -10000.0f;
                            sLog[m][nLoc] = val;
                        }
                    }
            }
            __syncthreads();
            // ---- online (flash) softmax update: 4 threads per row, 16 cols each ----
            {
                const int row = tid >> 2, sub = tid & 3;
                float lm = -3.0e38f;
#pragma unroll
                for (int j = 0; j < 16; ++j) lm = fmaxf(lm, sLog[row][sub * 16 + j]);
                float ls = 0.0f;
#pragma unroll
                for (int j = 0; j < 16; ++j) ls += __expf(sLog[row][sub * 16 + j] - lm);
                sPm[row][sub] = lm;
                sPl[row][sub] = ls;
            }
            __syncthreads();
            if (tid < BM) {
                float M = sRunM[tid], L = sRunL[tid];
#pragma unroll
                for (int s2 = 0; s2 < 4; ++s2) {
                    float m2 = sPm[tid][s2], l2 = sPl[tid][s2];
                    if (m2 > M) { L = L * __expf(M - m2) + l2; M = m2; }
                    else        { L += l2 * __expf(m2 - M); }
                }
                sRunM[tid] = M;
                sRunL[tid] = L;
            }
            __syncthreads();
        }
    }

    if (tid < BM) {
        size_t idx = ((size_t)(rowBase + tid) * NCHUNK + chunk) * 2;
        partials[idx + 0] = sRunM[tid];
        partials[idx + 1] = sRunL[tid];
    }
}

// ---------------- kernel 4: merge partials + true logit -> per-block NLL sums ----------------
__global__ __launch_bounds__(256) void row_lse_kernel(
    const float* __restrict__ true_logits, const float* __restrict__ partials,
    float* __restrict__ blockSums) {
    __shared__ float red[256];
    const int row = blockIdx.x * 256 + threadIdx.x;
    float t = true_logits[row];
    float M = t, L = 1.0f;                 // exp(t - M) = 1 : the true logit itself
#pragma unroll
    for (int c = 0; c < NCHUNK; ++c) {
        size_t idx = ((size_t)row * NCHUNK + c) * 2;
        float m2 = partials[idx + 0];
        float l2 = partials[idx + 1];
        if (m2 > M) { L = L * expf(M - m2) + l2; M = m2; }
        else        { L += l2 * expf(m2 - M); }
    }
    red[threadIdx.x] = (logf(L) + M) - t;  // nll contribution of this row
    __syncthreads();
    for (int off = 128; off > 0; off >>= 1) {
        if ((int)threadIdx.x < off) red[threadIdx.x] += red[threadIdx.x + off];
        __syncthreads();
    }
    if (threadIdx.x == 0) blockSums[blockIdx.x] = red[0];
}

__global__ void final_sum_kernel(const float* __restrict__ blockSums, float* __restrict__ out) {
    if (threadIdx.x == 0) {
        float s = 0.0f;
        for (int i = 0; i < B_DIM / 256; ++i) s += blockSums[i];
        out[0] = s;
    }
}

// ---------------- launch ----------------
extern "C" void kernel_launch(void* const* d_in, const int* in_sizes, int n_in,
                              void* d_out, int out_size, void* d_ws, size_t ws_size,
                              hipStream_t stream) {
    (void)n_in; (void)out_size; (void)ws_size;
    const float* emb         = (const float*)d_in[0];
    const int*   targets     = (const int*)d_in[1];
    const int*   sampled_ids = (const int*)d_in[2];
    const int*   num_tries   = (const int*)d_in[3];
    const float* w           = (const float*)d_in[4];
    const float* softmax_b   = (const float*)d_in[5];
    float*       out         = (float*)d_out;
    const int V = in_sizes[5];   // num_words = softmax_b length

    float* ws          = (float*)d_ws;
    float* colconst    = ws;                                   // S_DIM
    float* true_logits = colconst + S_DIM;                     // B_DIM
    float* partials    = true_logits + B_DIM;                  // B_DIM * NCHUNK * 2
    float* blockSums   = partials + (size_t)B_DIM * NCHUNK * 2;// B_DIM/256

    col_const_kernel<<<S_DIM / 256, 256, 0, stream>>>(sampled_ids, softmax_b, num_tries, V, colconst);
    true_logit_kernel<<<B_DIM, 256, 0, stream>>>(emb, targets, w, softmax_b, num_tries, V, true_logits);
    dim3 grid(NCHUNK, B_DIM / BM);   // 8 x 128
    gemm_lse_kernel<<<grid, 256, 0, stream>>>(emb, targets, sampled_ids, w, colconst, partials);
    row_lse_kernel<<<B_DIM / 256, 256, 0, stream>>>(true_logits, partials, blockSums);
    final_sum_kernel<<<1, 32, 0, stream>>>(blockSums, out);
}
